// Local_attention_4028679323866
// MI455X (gfx1250) — matrix operand order repaired and checked by reading, not compiled, not run on hardware
//
#include <hip/hip_runtime.h>
#include <hip/hip_bf16.h>
#include <stdint.h>

// Local attention block, MI455X (gfx1250), wave32 + WMMA + TDM.
// Requires ws_size >= ~202 MB.

#define BDIM 16
#define CDIM 512
#define NDIM 2048
#define C4D  128
#define NT16 (NDIM / 16)   // 128

typedef __attribute__((ext_vector_type(2)))  float  v2f;
typedef __attribute__((ext_vector_type(8)))  float  v8f;
typedef __attribute__((ext_vector_type(16))) __bf16 v16bf;
typedef __attribute__((ext_vector_type(4)))  unsigned int v4u;
typedef __attribute__((ext_vector_type(8)))  int    v8i;
typedef __attribute__((ext_vector_type(4)))  int    v4i;

#if __has_builtin(__builtin_amdgcn_tensor_load_to_lds)
#define HAVE_TDM 1
#else
#define HAVE_TDM 0
#endif

union BF { v16bf v; uint32_t u[8]; uint4 q[2]; };

__device__ __forceinline__ uint16_t f2bf(float f) {
  uint32_t u = __builtin_bit_cast(uint32_t, f);
  uint32_t r = u + 0x7FFFu + ((u >> 16) & 1u);
  return (uint16_t)(r >> 16);
}
__device__ __forceinline__ uint32_t pk2(float a, float b) {
  return (uint32_t)f2bf(a) | ((uint32_t)f2bf(b) << 16);
}

// ---------------- weight convert: wv, wt -> bf16 ----------------
__global__ void k_cvt_w(const float* __restrict__ wv, const float* __restrict__ wt,
                        uint16_t* __restrict__ wvb, uint16_t* __restrict__ wtb) {
  int i = blockIdx.x * 256 + threadIdx.x;   // 0 .. 512*512-1
  wvb[i] = f2bf(wv[i]);
  wtb[i] = f2bf(wt[i]);
}

// ---------------- x[b][C][N] -> xtb[b][N][C] (bf16, transposed) ----------------
__global__ void k_xt(const float* __restrict__ x, uint16_t* __restrict__ xtb) {
  __shared__ float tile[32][33];
  int t = threadIdx.x;
  const int nt = NDIM / 32, ct = CDIM / 32;           // 64, 16
  int bi = blockIdx.x;
  int b = bi / (nt * ct);
  int r = bi % (nt * ct);
  int ctile = r / nt, ntile = r % nt;
  const float* xb = x + (size_t)b * CDIM * NDIM;
  int col = t & 31, row0 = t >> 5;
#pragma unroll
  for (int i = 0; i < 4; ++i) {
    int cr = ctile * 32 + row0 + i * 8;
    tile[row0 + i * 8][col] = xb[(size_t)cr * NDIM + ntile * 32 + col];
  }
  __syncthreads();
  int nl = t >> 3;            // 0..31 (n within tile)
  int cg = (t & 7) * 4;       // group of 4 c
  uint32_t u0 = pk2(tile[cg + 0][nl], tile[cg + 1][nl]);
  uint32_t u1 = pk2(tile[cg + 2][nl], tile[cg + 3][nl]);
  uint16_t* op = xtb + (size_t)b * NDIM * CDIM + (size_t)(ntile * 32 + nl) * CDIM + ctile * 32 + cg;
  *(uint2*)op = make_uint2(u0, u1);
}

// ---------------- qk projection: XK[b][d][n], P[b][n][d]  (fp32 WMMA) ----------------
__global__ void k_qk(const float* __restrict__ x, const float* __restrict__ wqk,
                     float* __restrict__ XK, float* __restrict__ P) {
  int lane = threadIdx.x & 31, wv_ = threadIdx.x >> 5;
  int tile = blockIdx.x * 8 + wv_;               // b * 8 dtiles * 128 ntiles
  int b = tile / (8 * NT16);
  int r = tile % (8 * NT16);
  int d0 = (r / NT16) * 16, n0 = (r % NT16) * 16;
  const float* xb = x + (size_t)b * CDIM * NDIM;
  int l = lane & 15, h = lane >> 4;
  v8f acc = {};
#pragma unroll 4
  for (int c0 = 0; c0 < CDIM; c0 += 4) {
    v2f a = *(const v2f*)(wqk + (size_t)(d0 + l) * CDIM + c0 + 2 * h);
    v2f bf;
    bf.x = xb[(size_t)(c0 + 2 * h) * NDIM + n0 + l];
    bf.y = xb[(size_t)(c0 + 2 * h + 1) * NDIM + n0 + l];
    acc = __builtin_amdgcn_wmma_f32_16x16x4_f32(false, a, false, bf, (short)0, acc, false, false);
  }
  float* XKb = XK + (size_t)b * C4D * NDIM;
  float* Pb  = P  + (size_t)b * NDIM * C4D;
#pragma unroll
  for (int rr = 0; rr < 8; ++rr) {
    int dd = d0 + rr + 8 * h;
    XKb[(size_t)dd * NDIM + n0 + l] = acc[rr];
    Pb[(size_t)(n0 + l) * C4D + dd] = acc[rr];
  }
}

// ---------------- x_v = wv x + bv -> bf16  (bf16 WMMA) ----------------
__global__ void k_v(const uint16_t* __restrict__ xtb, const uint16_t* __restrict__ wvb,
                    const float* __restrict__ bv, uint16_t* __restrict__ xvb) {
  int lane = threadIdx.x & 31, w = threadIdx.x >> 5;
  int tile = blockIdx.x * 8 + w;                 // b * 32 ctiles * 128 ntiles
  int b = tile / (32 * NT16);
  int r = tile % (32 * NT16);
  int c0 = (r / NT16) * 16, n0 = (r % NT16) * 16;
  int l = lane & 15, h = lane >> 4;
  const uint16_t* xtB = xtb + (size_t)b * NDIM * CDIM;
  v8f acc = {};
#pragma unroll 2
  for (int k0 = 0; k0 < CDIM; k0 += 32) {
    BF a, bb;
    const uint16_t* ap = wvb + (size_t)(c0 + l) * CDIM + k0 + h * 16;
    a.q[0] = *(const uint4*)ap;
    a.q[1] = *(const uint4*)(ap + 8);
    const uint16_t* bp = xtB + (size_t)(n0 + l) * CDIM + k0 + h * 16;
    bb.q[0] = *(const uint4*)bp;
    bb.q[1] = *(const uint4*)(bp + 8);
    acc = __builtin_amdgcn_wmma_f32_16x16x32_bf16(false, a.v, false, bb.v, (short)0, acc, false, false);
  }
  uint16_t* outB = xvb + (size_t)b * CDIM * NDIM;
#pragma unroll
  for (int rr = 0; rr < 8; ++rr) {
    int cc = c0 + rr + 8 * h;
    outB[(size_t)cc * NDIM + n0 + l] = f2bf(acc[rr] + bv[cc]);
  }
}

// ---------------- softmax row stats: online max/sum over m  (fp32 WMMA) ----------------
__global__ void k_stats(const float* __restrict__ P, const float* __restrict__ XK,
                        float* __restrict__ rowmax, float* __restrict__ rowinv) {
  int lane = threadIdx.x & 31, w = threadIdx.x >> 5;
  int wid = blockIdx.x * 8 + w;                  // b * 128 n-blocks
  int b = wid / NT16;
  int n0 = (wid % NT16) * 16;
  int l = lane & 15, h = lane >> 4;
  const float* Pb  = P  + (size_t)b * NDIM * C4D;
  const float* XKb = XK + (size_t)b * C4D * NDIM;
  v2f afr[32];
#pragma unroll
  for (int kk = 0; kk < 32; ++kk)
    afr[kk] = *(const v2f*)(Pb + (size_t)(n0 + l) * C4D + kk * 4 + 2 * h);
  float m_run[8], s_run[8];
#pragma unroll
  for (int rr = 0; rr < 8; ++rr) { m_run[rr] = -3.0e38f; s_run[rr] = 0.f; }
  for (int m0 = 0; m0 < NDIM; m0 += 16) {
    v8f acc = {};
#pragma unroll 8
    for (int kk = 0; kk < 32; ++kk) {
      int d0 = kk * 4;
      v2f bf;
      bf.x = XKb[(size_t)(d0 + 2 * h) * NDIM + m0 + l];
      bf.y = XKb[(size_t)(d0 + 2 * h + 1) * NDIM + m0 + l];
      acc = __builtin_amdgcn_wmma_f32_16x16x4_f32(false, afr[kk], false, bf, (short)0, acc, false, false);
    }
#pragma unroll
    for (int rr = 0; rr < 8; ++rr) {
      float v = acc[rr];
      float tm = v;
      tm = fmaxf(tm, __shfl_xor(tm, 1, 32));
      tm = fmaxf(tm, __shfl_xor(tm, 2, 32));
      tm = fmaxf(tm, __shfl_xor(tm, 4, 32));
      tm = fmaxf(tm, __shfl_xor(tm, 8, 32));
      float nm = fmaxf(m_run[rr], tm);
      float e = __expf(v - nm);
      e += __shfl_xor(e, 1, 32);
      e += __shfl_xor(e, 2, 32);
      e += __shfl_xor(e, 4, 32);
      e += __shfl_xor(e, 8, 32);
      s_run[rr] = s_run[rr] * __expf(m_run[rr] - nm) + e;
      m_run[rr] = nm;
    }
  }
  if (l == 0) {
#pragma unroll
    for (int rr = 0; rr < 8; ++rr) {
      int row = n0 + rr + 8 * h;
      rowmax[(size_t)b * NDIM + row] = m_run[rr];
      rowinv[(size_t)b * NDIM + row] = 1.0f / s_run[rr];
    }
  }
}

// ---------------- fused energy -> weights -> x_r -> (x - x_r)^T bf16 ----------------
#define WTP 136   // padded n-pitch of weight tile (bf16): conflict-free b128 reads
__global__ void k_attn(const float* __restrict__ P, const float* __restrict__ XK,
                       const uint16_t* __restrict__ xvb, const float* __restrict__ x,
                       const float* __restrict__ rowmax, const float* __restrict__ rowinv,
                       uint16_t* __restrict__ dxrt) {
  __shared__ float    s_xk[C4D * 16];      // XK[:, m-block]  8KB
  __shared__ uint16_t s_wt[16 * WTP];      // W^T[m][n-chunk] 4.25KB
  __shared__ float    s_tr[8][16 * 17];    // per-wave transpose buffer 8.5KB
  int t = threadIdx.x, lane = t & 31, w = t >> 5;
  int b = blockIdx.x / NT16;
  int m0 = (blockIdx.x % NT16) * 16;
  int l = lane & 15, h = lane >> 4;
  const float*    Pb  = P   + (size_t)b * NDIM * C4D;
  const float*    XKb = XK  + (size_t)b * C4D * NDIM;
  const uint16_t* xvB = xvb + (size_t)b * CDIM * NDIM;
  const float*    rmB = rowmax + (size_t)b * NDIM;
  const float*    riB = rowinv + (size_t)b * NDIM;

#if HAVE_TDM
  // Stage XK[:, m0:m0+16] (128x16 dwords) into LDS with one Tensor-DMA op.
  // D# group0: count=1 | lds_addr | 57b global addr | type=2 ("image")
  // D# group1: data_size=4B, tensor_dim0=2048 (stride 2048), tensor_dim1=128,
  //            tile_dim0=16, tile_dim1=128  -> row-major contiguous into LDS.
  if (t < 32) {
    uint64_t ga  = (uint64_t)(uintptr_t)(XKb + m0);
    uint32_t lds = (uint32_t)(uintptr_t)&s_xk[0];
    v4u g0 = { 1u, lds, (uint32_t)ga,
               (uint32_t)((ga >> 32) & 0x01FFFFFFu) | (2u << 30) };
    v8i g1 = { (int)0x00020000,          // data_size = 2 (4 bytes)
               (int)0x08000000,          // tensor_dim0[15:0]=2048 in bits 63:48
               (int)0x00800000,          // tensor_dim1[15:0]=128 in bits 95:80
               (int)0x00100000,          // tile_dim0=16 in bits 127:112
               128,                      // tile_dim1=128 in bits 143:128
               2048,                     // tensor_dim0_stride[31:0]
               0, 0 };
    v4i gz = { 0, 0, 0, 0 };
    v8i gz8 = { 0, 0, 0, 0, 0, 0, 0, 0 };
    __builtin_amdgcn_tensor_load_to_lds(g0, g1, gz, gz, gz8, 0);
    __builtin_amdgcn_s_wait_tensorcnt(0);
  }
#else
#pragma unroll
  for (int i = 0; i < 8; ++i) {           // manual staging fallback
    int idx = i * 256 + t;
    s_xk[idx] = XKb[(size_t)(idx >> 4) * NDIM + m0 + (idx & 15)];
  }
#endif
  __syncthreads();

  v8f acc[4] = {};
  for (int nc = 0; nc < NDIM; nc += 128) {
    // step 1: wave w computes energy rows [nc+16w, nc+16w+16), applies softmax weights
    int nn = nc + w * 16;
    if (nc + 128 < NDIM)   // prefetch next chunk's P rows (global_prefetch_b8)
      __builtin_prefetch(Pb + (size_t)(nn + 128 + l) * C4D, 0, 3);
    v8f e = {};
#pragma unroll 8
    for (int kk = 0; kk < 32; ++kk) {
      int d0 = kk * 4;
      v2f a = *(const v2f*)(Pb + (size_t)(nn + l) * C4D + d0 + 2 * h);
      v2f bf;
      bf.x = s_xk[(d0 + 2 * h) * 16 + l];
      bf.y = s_xk[(d0 + 2 * h + 1) * 16 + l];
      e = __builtin_amdgcn_wmma_f32_16x16x4_f32(false, a, false, bf, (short)0, e, false, false);
    }
#pragma unroll
    for (int rr = 0; rr < 8; ++rr) {
      int row = nn + rr + 8 * h;
      float wgt = __expf(e[rr] - rmB[row]) * riB[row];
      s_wt[l * WTP + (w * 16 + rr + 8 * h)] = f2bf(wgt);
    }
    __syncthreads();
    // step 2: x_r accumulation, 4 c-tiles per wave, K = this 128-row n-chunk
#pragma unroll
    for (int i = 0; i < 4; ++i) {
      int c0 = (w * 4 + i) * 16;
#pragma unroll
      for (int kk = 0; kk < 4; ++kk) {
        BF a, bb;
        const uint16_t* ap = xvB + (size_t)(c0 + l) * NDIM + nc + kk * 32 + h * 16;
        a.q[0] = *(const uint4*)ap;
        a.q[1] = *(const uint4*)(ap + 8);
        const uint16_t* bp = s_wt + l * WTP + kk * 32 + h * 16;
        bb.q[0] = *(const uint4*)bp;
        bb.q[1] = *(const uint4*)(bp + 8);
        acc[i] = __builtin_amdgcn_wmma_f32_16x16x32_bf16(false, a.v, false, bb.v, (short)0, acc[i], false, false);
      }
    }
    __syncthreads();
  }
  // dxr = x - x_r, store transposed bf16 [N][C]
  const float* xB = x + (size_t)b * CDIM * NDIM;
  uint16_t* dxB = dxrt + (size_t)b * NDIM * CDIM;
#pragma unroll
  for (int i = 0; i < 4; ++i) {
    int c0 = (w * 4 + i) * 16;
#pragma unroll
    for (int rr = 0; rr < 8; ++rr) {
      int cl = rr + 8 * h;
      float dv = xB[(size_t)(c0 + cl) * NDIM + m0 + l] - acc[i][rr];
      s_tr[w][l * 17 + cl] = dv;
    }
    __syncthreads();
    uint32_t u0 = pk2(s_tr[w][l * 17 + h * 8 + 0], s_tr[w][l * 17 + h * 8 + 1]);
    uint32_t u1 = pk2(s_tr[w][l * 17 + h * 8 + 2], s_tr[w][l * 17 + h * 8 + 3]);
    uint32_t u2 = pk2(s_tr[w][l * 17 + h * 8 + 4], s_tr[w][l * 17 + h * 8 + 5]);
    uint32_t u3 = pk2(s_tr[w][l * 17 + h * 8 + 6], s_tr[w][l * 17 + h * 8 + 7]);
    uint16_t* op = dxB + (size_t)(m0 + l) * CDIM + c0 + h * 8;
    *(uint4*)op = make_uint4(u0, u1, u2, u3);
    __syncthreads();
  }
}

// ---------------- t = wt (x - x_r) + bt; deterministic BN partial sums ----------------
__global__ void k_trans(const uint16_t* __restrict__ dxrt, const uint16_t* __restrict__ wtb,
                        const float* __restrict__ bt, float* __restrict__ tout,
                        float* __restrict__ psum, float* __restrict__ psum2) {
  int lane = threadIdx.x & 31, w = threadIdx.x >> 5;
  int tile = blockIdx.x * 8 + w;                 // b * 32 ctiles * 128 mtiles
  int b = tile / (32 * NT16);
  int r = tile % (32 * NT16);
  int c0 = (r / NT16) * 16, m0 = (r % NT16) * 16;
  int l = lane & 15, h = lane >> 4;
  const uint16_t* dxB = dxrt + (size_t)b * NDIM * CDIM;
  v8f acc = {};
#pragma unroll 2
  for (int k0 = 0; k0 < CDIM; k0 += 32) {
    BF a, bb;
    const uint16_t* ap = wtb + (size_t)(c0 + l) * CDIM + k0 + h * 16;
    a.q[0] = *(const uint4*)ap;
    a.q[1] = *(const uint4*)(ap + 8);
    const uint16_t* bp = dxB + (size_t)(m0 + l) * CDIM + k0 + h * 16;
    bb.q[0] = *(const uint4*)bp;
    bb.q[1] = *(const uint4*)(bp + 8);
    acc = __builtin_amdgcn_wmma_f32_16x16x32_bf16(false, a.v, false, bb.v, (short)0, acc, false, false);
  }
  float* tB = tout + (size_t)b * CDIM * NDIM;
#pragma unroll
  for (int rr = 0; rr < 8; ++rr) {
    int cc = c0 + rr + 8 * h;
    float v = acc[rr] + bt[cc];
    tB[(size_t)cc * NDIM + m0 + l] = v;
    float s = v, s2 = v * v;
    s += __shfl_xor(s, 1, 32); s2 += __shfl_xor(s2, 1, 32);
    s += __shfl_xor(s, 2, 32); s2 += __shfl_xor(s2, 2, 32);
    s += __shfl_xor(s, 4, 32); s2 += __shfl_xor(s2, 4, 32);
    s += __shfl_xor(s, 8, 32); s2 += __shfl_xor(s2, 8, 32);
    if (l == 0) {
      size_t pi = (size_t)cc * (BDIM * NT16) + b * NT16 + (m0 >> 4);
      psum[pi] = s; psum2[pi] = s2;
    }
  }
}

// ---------------- fixed-order channel reduction ----------------
__global__ void k_red(const float* __restrict__ ps, const float* __restrict__ ps2,
                      float* __restrict__ sums, float* __restrict__ sumsq) {
  __shared__ float sh[256], sh2[256];
  int c = blockIdx.x, t = threadIdx.x;
  float a = 0.f, a2 = 0.f;
  for (int i = t; i < BDIM * NT16; i += 256) {
    a  += ps [(size_t)c * (BDIM * NT16) + i];
    a2 += ps2[(size_t)c * (BDIM * NT16) + i];
  }
  sh[t] = a; sh2[t] = a2;
  __syncthreads();
  for (int s = 128; s > 0; s >>= 1) {
    if (t < s) { sh[t] += sh[t + s]; sh2[t] += sh2[t + s]; }
    __syncthreads();
  }
  if (t == 0) { sums[c] = sh[0]; sumsq[c] = sh2[0]; }
}

// ---------------- BN (batch stats) + ReLU + residual ----------------
__global__ void k_final(const float* __restrict__ x, const float* __restrict__ tin,
                        const float* __restrict__ sums, const float* __restrict__ sumsq,
                        const float* __restrict__ gamma, const float* __restrict__ beta,
                        float* __restrict__ out) {
  size_t idx = (size_t)blockIdx.x * 256 + threadIdx.x;
  int c = (int)((idx / NDIM) % CDIM);
  const float inv = 1.0f / (float)(BDIM * NDIM);
  float mean = sums[c] * inv;
  float var  = sumsq[c] * inv - mean * mean;
  float rs   = rsqrtf(var + 1e-5f);
  float th   = (tin[idx] - mean) * rs * gamma[c] + beta[c];
  out[idx] = x[idx] + fmaxf(th, 0.f);
}

extern "C" void kernel_launch(void* const* d_in, const int* in_sizes, int n_in,
                              void* d_out, int out_size, void* d_ws, size_t ws_size,
                              hipStream_t stream) {
  (void)in_sizes; (void)n_in; (void)out_size; (void)ws_size;
  const float* x     = (const float*)d_in[0];
  const float* wqk   = (const float*)d_in[1];
  const float* wvw   = (const float*)d_in[2];
  const float* bv    = (const float*)d_in[3];
  const float* wtw   = (const float*)d_in[4];
  const float* btp   = (const float*)d_in[5];
  const float* gamma = (const float*)d_in[6];
  const float* beta  = (const float*)d_in[7];
  float* out = (float*)d_out;

  char* ws = (char*)d_ws;
  const size_t MB = 1u << 20;
  float*    XK     = (float*)(ws);                          // 16 MB
  float*    P      = (float*)(ws + 16 * MB);                // 16 MB
  uint16_t* xtb    = (uint16_t*)(ws + 32 * MB);             // 32 MB
  uint16_t* xvb    = (uint16_t*)(ws + 64 * MB);             // 32 MB
  uint16_t* dxrt   = (uint16_t*)(ws + 96 * MB);             // 32 MB
  float*    tbuf   = (float*)(ws + 128 * MB);               // 64 MB
  float*    rowmax = (float*)(ws + 192 * MB);               // 128 KB
  float*    rowinv = (float*)(ws + 192 * MB + 128 * 1024);  // 128 KB
  uint16_t* wvb    = (uint16_t*)(ws + 192 * MB + 256 * 1024);
  uint16_t* wtb    = (uint16_t*)(ws + 192 * MB + 768 * 1024);
  float*    sums   = (float*)(ws + 193 * MB);               // 4 KB
  float*    sumsq  = sums + CDIM;
  float*    psum   = (float*)(ws + 194 * MB);               // 4 MB
  float*    psum2  = (float*)(ws + 198 * MB);               // 4 MB

  k_cvt_w<<<(CDIM * CDIM) / 256, 256, 0, stream>>>(wvw, wtw, wvb, wtb);
  k_xt<<<BDIM * (CDIM / 32) * (NDIM / 32), 256, 0, stream>>>(x, xtb);
  k_qk<<<BDIM * 8 * NT16 / 8, 256, 0, stream>>>(x, wqk, XK, P);
  k_v<<<BDIM * 32 * NT16 / 8, 256, 0, stream>>>(xtb, wvb, bv, xvb);
  k_stats<<<BDIM * NT16 / 8, 256, 0, stream>>>(P, XK, rowmax, rowinv);
  k_attn<<<BDIM * NT16, 256, 0, stream>>>(P, XK, xvb, x, rowmax, rowinv, dxrt);
  k_trans<<<BDIM * 32 * NT16 / 8, 256, 0, stream>>>(dxrt, wtb, btp, tbuf, psum, psum2);
  k_red<<<CDIM, 256, 0, stream>>>(psum, psum2, sums, sumsq);
  k_final<<<(size_t)BDIM * CDIM * NDIM / 256, 256, 0, stream>>>(x, tbuf, sums, sumsq, gamma, beta, out);
}
